// ColorHistograms_43593918054920
// MI455X (gfx1250) — compile-verified, matrix operand order checked by reading, not run on hardware
//
#include <hip/hip_runtime.h>
#include <hip/hip_bf16.h>

#define LOOKUP_WINDOW 101
#define OUTPUT_DIM    128
#define NUM_BINS      512
#define B_DIM         16
#define T_DIM         100
#define T_PAD         112          // 7 tiles of 16
#define W_PAD         128          // 101 window taps padded to 8 K-chunks of 16
#define PIXELS        (27 * 48)    // 1296
#define K_DIM         NUM_BINS     // 512

typedef __attribute__((ext_vector_type(16))) _Float16 v16h;
typedef __attribute__((ext_vector_type(8)))  _Float16 v8h;
typedef __attribute__((ext_vector_type(8)))  float    v8f;

// ---------------------------------------------------------------------------
// Kernel A: per-frame 512-bin color histogram -> L2 normalize -> f16 row.
// One block per (b, t_padded). Pad rows (t >= 100) are written as zeros so the
// downstream WMMA GEMM naturally produces zero similarity in the pad region.
// ---------------------------------------------------------------------------
__global__ __launch_bounds__(256) void hist_norm_kernel(const int* __restrict__ frames,
                                                        _Float16* __restrict__ xh) {
    __shared__ int   hist[NUM_BINS];
    __shared__ float red[256];

    const int blk = blockIdx.x;          // 16 * 112
    const int b   = blk / T_PAD;
    const int t   = blk % T_PAD;
    const int tid = threadIdx.x;

    hist[tid]       = 0;
    hist[tid + 256] = 0;
    __syncthreads();

    if (t < T_DIM) {
        const int* f = frames + (size_t)(b * T_DIM + t) * PIXELS * 3;
        for (int p = tid; p < PIXELS; p += 256) {
            const int r  = f[p * 3 + 0];
            const int g  = f[p * 3 + 1];
            const int bl = f[p * 3 + 2];
            const int bin = ((r >> 5) << 6) | ((g >> 5) << 3) | (bl >> 5);
            atomicAdd(&hist[bin], 1);    // ds_add_u32
        }
    }
    __syncthreads();

    const int c0 = hist[tid];
    const int c1 = hist[tid + 256];
    red[tid] = (float)c0 * (float)c0 + (float)c1 * (float)c1;
    __syncthreads();
    for (int s = 128; s > 0; s >>= 1) {
        if (tid < s) red[tid] += red[tid + s];
        __syncthreads();
    }

    const float inv = (t < T_DIM) ? __frsqrt_rn(red[0] > 0.0f ? red[0] : 1.0f) : 0.0f;

    _Float16* row = xh + ((size_t)b * T_PAD + t) * (size_t)K_DIM;
    row[tid]       = (_Float16)((float)c0 * inv);
    row[tid + 256] = (_Float16)((float)c1 * inv);
}

// ---------------------------------------------------------------------------
// Kernel B: sim[b] = X[b] * X[b]^T via v_wmma_f32_16x16x32_f16.
// One wave (32 lanes, EXEC all-ones) per 16x16 output tile; 16 WMMAs over
// K=512. Because C = X * X^T, the B-operand tile (column-major slice of X^T)
// has exactly the same per-lane VGPR layout as the A-operand tile:
//   lane l: row = base + (l & 15); hi8 = (l >= 16) ? 8 : 0
//   per 32-K chunk: halves [k0+hi8 .. +7] -> VGPR0..3, [k0+16+hi8 .. +7] -> VGPR4..7
// C layout: VGPR r, lane l -> (M = r + (l>=16 ? 8 : 0), N = l & 15).
// ---------------------------------------------------------------------------
__global__ __launch_bounds__(32) void sim_wmma_kernel(const _Float16* __restrict__ xh,
                                                      float* __restrict__ sim) {
    const int blk  = blockIdx.x;         // 16 * 49
    const int b    = blk / 49;
    const int tij  = blk % 49;
    const int ti   = tij / 7;
    const int tj   = tij % 7;
    const int lane = threadIdx.x;
    const int lo   = lane & 15;
    const int hi8  = (lane >> 4) * 8;    // 0 for lanes 0-15, 8 for lanes 16-31

    const _Float16* xb   = xh + (size_t)b * T_PAD * K_DIM;
    const _Float16* rowA = xb + (size_t)(ti * 16 + lo) * K_DIM;
    const _Float16* rowB = xb + (size_t)(tj * 16 + lo) * K_DIM;

    v8f c = {};
    #pragma unroll
    for (int k0 = 0; k0 < K_DIM; k0 += 32) {
        __builtin_prefetch(rowA + k0 + 128, 0, 1);   // global_prefetch_b8
        const v8h a0 = *(const v8h*)(rowA + k0 + hi8);
        const v8h a1 = *(const v8h*)(rowA + k0 + 16 + hi8);
        const v8h b0 = *(const v8h*)(rowB + k0 + hi8);
        const v8h b1 = *(const v8h*)(rowB + k0 + 16 + hi8);
        const v16h av = __builtin_shufflevector(a0, a1, 0,1,2,3,4,5,6,7,8,9,10,11,12,13,14,15);
        const v16h bv = __builtin_shufflevector(b0, b1, 0,1,2,3,4,5,6,7,8,9,10,11,12,13,14,15);
        c = __builtin_amdgcn_wmma_f32_16x16x32_f16(
                /*neg_a=*/false, av, /*neg_b=*/false, bv,
                /*c_mod=*/(short)0, c, /*reuse_a=*/false, /*reuse_b=*/false);
    }

    const int   mbase = ti * 16 + (hi8 & 8);   // +8 for upper half-wave
    const int   n     = tj * 16 + lo;
    float*      srow  = sim + (size_t)b * T_PAD * T_PAD;
    #pragma unroll
    for (int r = 0; r < 8; ++r) {
        srow[(size_t)(mbase + r) * T_PAD + n] = c[r];
    }
}

// ---------------------------------------------------------------------------
// Kernel C1: gather the banded diagonal of sim into a dense f16 matrix
// bandh[b][t][w] = sim[b][t][t + w - 50]  (zero outside [0,100), w >= 101, or
// pad rows t >= 100). This turns the FC into a clean dense WMMA GEMM.
// ---------------------------------------------------------------------------
__global__ __launch_bounds__(W_PAD) void band_build_kernel(const float* __restrict__ sim,
                                                           _Float16* __restrict__ bandh) {
    const int blk = blockIdx.x;          // 16 * 112
    const int b   = blk / T_PAD;
    const int t   = blk % T_PAD;
    const int w   = threadIdx.x;         // 0..127

    float val = 0.0f;
    if (t < T_DIM && w < LOOKUP_WINDOW) {
        const int s = t + w - (LOOKUP_WINDOW - 1) / 2;
        if (s >= 0 && s < T_DIM) {
            val = sim[((size_t)b * T_PAD + t) * T_PAD + s];
        }
    }
    bandh[((size_t)b * T_PAD + t) * W_PAD + w] = (_Float16)val;
}

// ---------------------------------------------------------------------------
// Kernel C2: transpose fc_w (101,128) -> f16 fcw_t[n][w] with w padded to 128.
// N-major storage makes the WMMA B-operand per-lane loads contiguous 16-byte
// v8h loads (same pattern as the A operand).
// ---------------------------------------------------------------------------
__global__ __launch_bounds__(W_PAD) void fcw_transpose_kernel(const float* __restrict__ fc_w,
                                                              _Float16* __restrict__ fcw_t) {
    const int n = blockIdx.x;            // 0..127 output channel
    const int w = threadIdx.x;           // 0..127 padded tap index
    const float v = (w < LOOKUP_WINDOW) ? fc_w[w * OUTPUT_DIM + n] : 0.0f;
    fcw_t[(size_t)n * W_PAD + w] = (_Float16)v;
}

// ---------------------------------------------------------------------------
// Kernel C3: out[b] = relu(bandh[b] (112x128) * fcw_t^T (128x128) + fc_b)
// via v_wmma_f32_16x16x32_f16. One wave per 16x16 output tile:
// 16 batches x 7 M-tiles x 8 N-tiles = 896 waves, K = 128 -> 4 WMMAs each.
// Fused bias + ReLU epilogue in the C layout; stores guarded to M < 100.
// ---------------------------------------------------------------------------
__global__ __launch_bounds__(32) void fc_wmma_kernel(const _Float16* __restrict__ bandh,
                                                     const _Float16* __restrict__ fcw_t,
                                                     const float* __restrict__ fc_b,
                                                     float* __restrict__ out) {
    const int blk  = blockIdx.x;         // 16 * 7 * 8
    const int b    = blk / 56;
    const int mn   = blk % 56;
    const int mt   = mn / 8;             // 0..6   M tile (time)
    const int nt   = mn % 8;             // 0..7   N tile (channel)
    const int lane = threadIdx.x;
    const int lo   = lane & 15;
    const int hi8  = (lane >> 4) * 8;

    const _Float16* rowA = bandh + ((size_t)b * T_PAD + mt * 16 + lo) * W_PAD;
    const _Float16* rowB = fcw_t + (size_t)(nt * 16 + lo) * W_PAD;

    v8f c = {};
    #pragma unroll
    for (int k0 = 0; k0 < W_PAD; k0 += 32) {
        const v8h a0 = *(const v8h*)(rowA + k0 + hi8);
        const v8h a1 = *(const v8h*)(rowA + k0 + 16 + hi8);
        const v8h b0 = *(const v8h*)(rowB + k0 + hi8);
        const v8h b1 = *(const v8h*)(rowB + k0 + 16 + hi8);
        const v16h av = __builtin_shufflevector(a0, a1, 0,1,2,3,4,5,6,7,8,9,10,11,12,13,14,15);
        const v16h bv = __builtin_shufflevector(b0, b1, 0,1,2,3,4,5,6,7,8,9,10,11,12,13,14,15);
        c = __builtin_amdgcn_wmma_f32_16x16x32_f16(
                false, av, false, bv, (short)0, c, false, false);
    }

    const int   n    = nt * 16 + lo;
    const float bias = fc_b[n];
    const int   mb   = mt * 16 + (hi8 & 8);
    #pragma unroll
    for (int r = 0; r < 8; ++r) {
        const int m = mb + r;
        if (m < T_DIM) {
            out[((size_t)b * T_DIM + m) * OUTPUT_DIM + n] = fmaxf(c[r] + bias, 0.0f);
        }
    }
}

// ---------------------------------------------------------------------------
extern "C" void kernel_launch(void* const* d_in, const int* in_sizes, int n_in,
                              void* d_out, int out_size, void* d_ws, size_t ws_size,
                              hipStream_t stream) {
    const int*   frames = (const int*)d_in[0];                 // (16,100,27,48,3) int32
    const float* fc_w   = (const float*)d_in[1];               // (101,128) f32
    const float* fc_b   = (const float*)d_in[2];               // (128,)    f32
    float*       out    = (float*)d_out;                       // (16,100,128) f32

    // Workspace layout (all 256B aligned):
    //   xh    f16 [16][112][512]  1,835,008 B @ 0
    //   sim   f32 [16][112][112]    802,816 B @ 1,835,008
    //   bandh f16 [16][112][128]    458,752 B @ 2,637,824
    //   fcw_t f16 [128][128]         32,768 B @ 3,096,576
    const size_t xh_bytes    = (size_t)B_DIM * T_PAD * K_DIM * sizeof(_Float16);
    const size_t sim_bytes   = (size_t)B_DIM * T_PAD * T_PAD * sizeof(float);
    const size_t bandh_bytes = (size_t)B_DIM * T_PAD * W_PAD * sizeof(_Float16);

    _Float16* xh    = (_Float16*)d_ws;
    float*    sim   = (float*)((char*)d_ws + xh_bytes);
    _Float16* bandh = (_Float16*)((char*)d_ws + xh_bytes + sim_bytes);
    _Float16* fcw_t = (_Float16*)((char*)d_ws + xh_bytes + sim_bytes + bandh_bytes);

    // Independent of the main chain -> launch first to overlap.
    fcw_transpose_kernel<<<OUTPUT_DIM, W_PAD, 0, stream>>>(fc_w, fcw_t);

    hist_norm_kernel<<<B_DIM * T_PAD, 256, 0, stream>>>(frames, xh);
    sim_wmma_kernel<<<B_DIM * 49, 32, 0, stream>>>(xh, sim);
    band_build_kernel<<<B_DIM * T_PAD, W_PAD, 0, stream>>>(sim, bandh);
    fc_wmma_kernel<<<B_DIM * 7 * 8, 32, 0, stream>>>(bandh, fcw_t, fc_b, out);
}
